// TinyGPT_67585605370649
// MI455X (gfx1250) — compile-verified, matrix operand order checked by reading, not run on hardware
//
#include <hip/hip_runtime.h>
#include <hip/hip_bf16.h>
#include <math.h>

// ---------------- model constants ----------------
#define Bc   2
#define Tc   1024
#define Dc   1024
#define Hc   16
#define Lc   8
#define DFFc 4096
#define Vc   32000
#define Nc   (Bc * Tc)          // 2048 tokens
#define DHc  64                 // head dim
#define QKVc (3 * Dc)           // 3072

typedef __attribute__((ext_vector_type(16))) __bf16 v16bf;
typedef __attribute__((ext_vector_type(8)))  __bf16 v8bf;
typedef __attribute__((ext_vector_type(8)))  float  v8f;

// CDNA5 async global->LDS copy (ASYNCcnt-tracked), if the toolchain exposes it.
// Probed signature: void(v4i addrspace(1)* src, v4i addrspace(3)* dst, Ii off, Ii cpol)
#if defined(__has_builtin)
#if __has_builtin(__builtin_amdgcn_global_load_async_to_lds_b128)
#define HAVE_ASYNC_LDS 1
#endif
#endif

#ifdef HAVE_ASYNC_LDS
typedef int v4i __attribute__((vector_size(16)));
typedef __attribute__((address_space(1))) v4i* as1_v4i;
typedef __attribute__((address_space(3))) v4i* as3_v4i;
#define GLD_ASYNC_B128(gp, lp) \
  __builtin_amdgcn_global_load_async_to_lds_b128((as1_v4i)(gp), (as3_v4i)(lp), 0, 0)
#endif

// ---------------------------------------------------------------------------
// Fragment loaders (CDNA5 WMMA 16x16x32 bf16 layouts, ISA 7.12.2):
//   A 16x32: lane L holds row L%16; half h=L/16 holds K in two contiguous
//   8-element chunks at K = h*8 and K = 16 + h*8.
//   B 32x16 (for C = A * W^T with row-major W[M][K]) mirrors A over W's rows.
// ---------------------------------------------------------------------------
__device__ inline v16bf frag_from_bf16(const __bf16* rowp, int hf) {
  const int ka = hf * 8, kb = 16 + hf * 8;
  v8bf a = *(const v8bf*)(rowp + ka);
  v8bf b = *(const v8bf*)(rowp + kb);
  v16bf r;
#pragma unroll
  for (int i = 0; i < 8; ++i) { r[i] = a[i]; r[i + 8] = b[i]; }
  return r;
}

__device__ inline v16bf frag_from_f32(const float* rowp, int hf) {
  const float4* pa = (const float4*)(rowp + hf * 8);
  const float4* pb = (const float4*)(rowp + 16 + hf * 8);
  float4 a0 = pa[0], a1 = pa[1], b0 = pb[0], b1 = pb[1];
  v16bf r;
  r[0]  = (__bf16)a0.x; r[1]  = (__bf16)a0.y; r[2]  = (__bf16)a0.z; r[3]  = (__bf16)a0.w;
  r[4]  = (__bf16)a1.x; r[5]  = (__bf16)a1.y; r[6]  = (__bf16)a1.z; r[7]  = (__bf16)a1.w;
  r[8]  = (__bf16)b0.x; r[9]  = (__bf16)b0.y; r[10] = (__bf16)b0.z; r[11] = (__bf16)b0.w;
  r[12] = (__bf16)b1.x; r[13] = (__bf16)b1.y; r[14] = (__bf16)b1.z; r[15] = (__bf16)b1.w;
  return r;
}

#define WMMA_BF16(A, B, C) \
  __builtin_amdgcn_wmma_f32_16x16x32_bf16(false, (A), false, (B), (short)0, (C), false, false)

// width-16 cross-lane reductions (rows of a WMMA C tile live on 16 lanes)
__device__ inline float rmax16(float v) {
  v = fmaxf(v, __shfl_xor(v, 1, 16));
  v = fmaxf(v, __shfl_xor(v, 2, 16));
  v = fmaxf(v, __shfl_xor(v, 4, 16));
  v = fmaxf(v, __shfl_xor(v, 8, 16));
  return v;
}
__device__ inline float rsum16(float v) {
  v += __shfl_xor(v, 1, 16);
  v += __shfl_xor(v, 2, 16);
  v += __shfl_xor(v, 4, 16);
  v += __shfl_xor(v, 8, 16);
  return v;
}

// ---------------------------------------------------------------------------
// GEMM: C[N x M] = A[N x K](bf16) * W[M x K](f32)^T  (+bias)(+GELU)(+res)
// 256 threads = 8 waves. Block tile 128x64; wave (wy in 0..3, wx in 0..1)
// owns a 32x32 sub-tile (2 A-frags x 2 B-frags -> 4 WMMA per K-step).
// A (128x32 bf16, 8KB) and W (64x32 f32, 8KB) tiles are staged into
// double-buffered LDS via ASYNCcnt-tracked global_load_async_to_lds_b128,
// prefetching K-step k+32 while WMMAs consume k.
// All problem dims are multiples of 128/64/32 -> no bounds checks.
// ---------------------------------------------------------------------------
__device__ inline void stage_tiles(const __bf16* Abase, const float* Wbase, int lda,
                                   __bf16* sAp, float* sWp, int tid) {
#pragma unroll
  for (int i = 0; i < 2; ++i) {
    const int c = tid + i * 256;
    const int ar = c >> 2, akc = c & 3;   // A: 128 rows x 4 chunks of 8 bf16
    const int wr = c >> 3, wkc = c & 7;   // W:  64 rows x 8 chunks of 4 f32
#ifdef HAVE_ASYNC_LDS
    GLD_ASYNC_B128(Abase + (size_t)ar * lda + akc * 8, &sAp[ar * 32 + akc * 8]);
    GLD_ASYNC_B128(Wbase + (size_t)wr * lda + wkc * 4, &sWp[wr * 32 + wkc * 4]);
#else
    *(v8bf*)&sAp[ar * 32 + akc * 8] = *(const v8bf*)(Abase + (size_t)ar * lda + akc * 8);
    *(float4*)&sWp[wr * 32 + wkc * 4] = *(const float4*)(Wbase + (size_t)wr * lda + wkc * 4);
#endif
  }
}

__global__ __launch_bounds__(256)
void gemm_bf16_wmma(const __bf16* __restrict__ A, const float* __restrict__ W,
                    const float* __restrict__ bias, const float* __restrict__ res,
                    float* __restrict__ out32, __bf16* __restrict__ out16,
                    int Kdim, int M, int gelu) {
  __shared__ __attribute__((aligned(16))) __bf16 sA[2][128 * 32]; // 16 KB
  __shared__ __attribute__((aligned(16))) float  sW[2][64 * 32];  // 16 KB

  const int tid  = threadIdx.x;
  const int lane = tid & 31;
  const int wave = tid >> 5;
  const int wy = wave & 3, wx = wave >> 2;
  const int hf = lane >> 4, ln = lane & 15;
  const size_t rb = (size_t)blockIdx.y * 128;   // block row base (N dim)
  const int    cb = blockIdx.x * 64;            // block col base (M dim)

  const __bf16* Ablk = A + rb * Kdim;
  const float*  Wblk = W + (size_t)cb * Kdim;

  v8f acc00 = {}, acc01 = {}, acc10 = {}, acc11 = {};

  const int nk = Kdim >> 5;
#ifdef HAVE_ASYNC_LDS
  stage_tiles(Ablk, Wblk, Kdim, sA[0], sW[0], tid);   // prologue: tile 0 in flight
#endif
  for (int kidx = 0; kidx < nk; ++kidx) {
    const int k = kidx << 5;
    const int p = kidx & 1;
    const bool lastIter = (kidx == nk - 1);
#ifdef HAVE_ASYNC_LDS
    if (!lastIter) {
      stage_tiles(Ablk + k + 32, Wblk + k + 32, Kdim, sA[p ^ 1], sW[p ^ 1], tid);
      // 4 async issues/thread/tile, in-order completion: <=4 outstanding
      // guarantees the current tile has landed while the next stays in flight.
      asm volatile("s_wait_asynccnt 0x4" ::: "memory");
    } else {
      asm volatile("s_wait_asynccnt 0x0" ::: "memory");
    }
#else
    stage_tiles(Ablk + k, Wblk + k, Kdim, sA[p], sW[p], tid);
#endif
    __syncthreads();

    v16bf af0 = frag_from_bf16(&sA[p][(wy * 32 + 0  + ln) * 32], hf);
    v16bf af1 = frag_from_bf16(&sA[p][(wy * 32 + 16 + ln) * 32], hf);
    v16bf bf0 = frag_from_f32(&sW[p][(wx * 32 + 0  + ln) * 32], hf);
    v16bf bf1 = frag_from_f32(&sW[p][(wx * 32 + 16 + ln) * 32], hf);
    acc00 = WMMA_BF16(af0, bf0, acc00);
    acc01 = WMMA_BF16(af0, bf1, acc01);
    acc10 = WMMA_BF16(af1, bf0, acc10);
    acc11 = WMMA_BF16(af1, bf1, acc11);

    __syncthreads();   // all reads of buffer p done before it is refilled
  }

  const int c0 = cb + wx * 32;
  const float bias0 = bias ? bias[c0 + ln]      : 0.f;
  const float bias1 = bias ? bias[c0 + 16 + ln] : 0.f;
#pragma unroll
  for (int s = 0; s < 2; ++s) {
    const v8f a0 = s ? acc10 : acc00;
    const v8f a1 = s ? acc11 : acc01;
#pragma unroll
    for (int r = 0; r < 8; ++r) {
      const size_t row = rb + wy * 32 + s * 16 + r + 8 * hf;
      float v0 = a0[r] + bias0;
      float v1 = a1[r] + bias1;
      if (gelu) {
        v0 = 0.5f * v0 * (1.f + erff(v0 * 0.70710678118654752f));
        v1 = 0.5f * v1 * (1.f + erff(v1 * 0.70710678118654752f));
      }
      const size_t i0 = row * (size_t)M + (c0 + ln);
      const size_t i1 = row * (size_t)M + (c0 + 16 + ln);
      if (res)   { v0 += res[i0]; v1 += res[i1]; }
      if (out32) { out32[i0] = v0; out32[i1] = v1; }
      if (out16) { out16[i0] = (__bf16)v0; out16[i1] = (__bf16)v1; }
    }
  }
}

// ---------------------------------------------------------------------------
// Flash attention: one wave per (b, h, 16 query rows). Q fragments live in
// registers; K/V streamed 32 keys at a time; online softmax; P goes through
// a 1KB LDS tile to be re-read in A-fragment layout; O accumulated via WMMA.
// ---------------------------------------------------------------------------
__global__ __launch_bounds__(32)
void attn_kernel(const __bf16* __restrict__ qkv, __bf16* __restrict__ y) {
  __shared__ __attribute__((aligned(16))) __bf16 sP[16 * 32];

  const int lane = threadIdx.x;
  const int hf = lane >> 4, ln = lane & 15;
  const int T16 = Tc / 16;
  const int qt = blockIdx.x % T16;
  const int bh = blockIdx.x / T16;
  const int h = bh % Hc, b = bh / Hc;
  const int q0 = qt * 16;

  const __bf16* base = qkv + (size_t)b * Tc * QKVc;

  const __bf16* Qp = base + (size_t)(q0 + ln) * QKVc + h * DHc;
  const v16bf qf0 = frag_from_bf16(Qp + 0, hf);
  const v16bf qf1 = frag_from_bf16(Qp + 32, hf);

  v8f o[4] = {};
  float mi[8], li[8];
#pragma unroll
  for (int r = 0; r < 8; ++r) { mi[r] = -1e30f; li[r] = 0.f; }

  const int nblk = (q0 + 16 + 31) >> 5;
  for (int kb = 0; kb < nblk; ++kb) {
    const int ks = kb * 32;
    float p[2][8];
    float bm[8];
#pragma unroll
    for (int r = 0; r < 8; ++r) bm[r] = -1e30f;

    // S = Q K^T for 2 x 16 keys, scaled + causally masked
    for (int j = 0; j < 2; ++j) {
      const int kt0 = ks + j * 16;
      const __bf16* Kp = base + (size_t)(kt0 + ln) * QKVc + Dc + h * DHc;
      v16bf kf0 = frag_from_bf16(Kp + 0, hf);
      v16bf kf1 = frag_from_bf16(Kp + 32, hf);
      v8f s = {};
      s = WMMA_BF16(qf0, kf0, s);
      s = WMMA_BF16(qf1, kf1, s);
      const int key = kt0 + ln;
#pragma unroll
      for (int r = 0; r < 8; ++r) {
        const int qrow = q0 + r + 8 * hf;
        const float v = (key <= qrow) ? s[r] * 0.125f : -1e30f;
        p[j][r] = v;
        bm[r] = fmaxf(bm[r], rmax16(v));
      }
    }

    // online softmax update
#pragma unroll
    for (int r = 0; r < 8; ++r) {
      const float mn = fmaxf(mi[r], bm[r]);
      const float alpha = expf(mi[r] - mn);
      p[0][r] = expf(p[0][r] - mn);
      p[1][r] = expf(p[1][r] - mn);
      const float rowsum = rsum16(p[0][r]) + rsum16(p[1][r]);
      li[r] = li[r] * alpha + rowsum;
      mi[r] = mn;
      o[0][r] *= alpha; o[1][r] *= alpha; o[2][r] *= alpha; o[3][r] *= alpha;
    }

    // transpose P (C layout) -> A-fragment layout via LDS
    __syncthreads();
#pragma unroll
    for (int j = 0; j < 2; ++j)
#pragma unroll
      for (int r = 0; r < 8; ++r)
        sP[(r + 8 * hf) * 32 + j * 16 + ln] = (__bf16)p[j][r];
    __syncthreads();
    const v16bf pf = frag_from_bf16(&sP[ln * 32], hf);

    // O += P * V  (V streamed strided from qkv)
    const __bf16* Vb = base + (size_t)ks * QKVc + 2 * Dc + h * DHc;
    const int ka = hf * 8, kb2 = 16 + hf * 8;
#pragma unroll
    for (int c = 0; c < 4; ++c) {
      const __bf16* vp = Vb + c * 16 + ln;
      v16bf vf;
#pragma unroll
      for (int i = 0; i < 8; ++i) {
        vf[i]     = vp[(size_t)(ka  + i) * QKVc];
        vf[i + 8] = vp[(size_t)(kb2 + i) * QKVc];
      }
      o[c] = WMMA_BF16(pf, vf, o[c]);
    }
  }

#pragma unroll
  for (int c = 0; c < 4; ++c)
#pragma unroll
    for (int r = 0; r < 8; ++r) {
      const int qrow = q0 + r + 8 * hf;
      y[((size_t)b * Tc + qrow) * Dc + h * DHc + c * 16 + ln] =
          (__bf16)(o[c][r] / li[r]);
    }
}

// ---------------------------------------------------------------------------
// LayerNorm over D=1024, fp32 in -> bf16 out. One block per row.
// ---------------------------------------------------------------------------
__global__ __launch_bounds__(256)
void ln_kernel(const float* __restrict__ x, const float* __restrict__ g,
               const float* __restrict__ bta, __bf16* __restrict__ out) {
  __shared__ float red[256];
  const int n = blockIdx.x, t = threadIdx.x;
  const float* row = x + (size_t)n * Dc;
  float v[4], s = 0.f;
#pragma unroll
  for (int i = 0; i < 4; ++i) { v[i] = row[t + 256 * i]; s += v[i]; }
  red[t] = s; __syncthreads();
  for (int off = 128; off > 0; off >>= 1) {
    if (t < off) red[t] += red[t + off];
    __syncthreads();
  }
  const float mu = red[0] * (1.f / Dc);
  __syncthreads();
  float s2 = 0.f;
#pragma unroll
  for (int i = 0; i < 4; ++i) { const float d = v[i] - mu; s2 += d * d; }
  red[t] = s2; __syncthreads();
  for (int off = 128; off > 0; off >>= 1) {
    if (t < off) red[t] += red[t + off];
    __syncthreads();
  }
  const float rs = rsqrtf(red[0] * (1.f / Dc) + 1e-5f);
#pragma unroll
  for (int i = 0; i < 4; ++i) {
    const int c = t + 256 * i;
    out[(size_t)n * Dc + c] = (__bf16)((v[i] - mu) * rs * g[c] + bta[c]);
  }
}

// ---------------------------------------------------------------------------
// x = tok_emb[ids] + pos_emb[t]
// ---------------------------------------------------------------------------
__global__ __launch_bounds__(256)
void embed_kernel(const int* __restrict__ ids, const float* __restrict__ tok,
                  const float* __restrict__ pos, float* __restrict__ x) {
  const int n = blockIdx.x, t = threadIdx.x;
  const int id = ids[n];
  const int trow = n % Tc;
#pragma unroll
  for (int i = 0; i < 4; ++i) {
    const int c = t + 256 * i;
    x[(size_t)n * Dc + c] = tok[(size_t)id * Dc + c] + pos[(size_t)trow * Dc + c];
  }
}

// ---------------------------------------------------------------------------
// mean NLL from logits; loss accumulated via atomicAdd into *loss.
// ---------------------------------------------------------------------------
__global__ void zero_kernel(float* p) { *p = 0.f; }

__global__ __launch_bounds__(256)
void loss_kernel(const float* __restrict__ logits, const int* __restrict__ targets,
                 float* __restrict__ loss) {
  __shared__ float red[256];
  const int n = blockIdx.x, t = threadIdx.x;
  const float* row = logits + (size_t)n * Vc;
  float mx = -3.4e38f;
  for (int c = t; c < Vc; c += 256) mx = fmaxf(mx, row[c]);
  red[t] = mx; __syncthreads();
  for (int off = 128; off > 0; off >>= 1) {
    if (t < off) red[t] = fmaxf(red[t], red[t + off]);
    __syncthreads();
  }
  mx = red[0]; __syncthreads();
  float s = 0.f;
  for (int c = t; c < Vc; c += 256) s += expf(row[c] - mx);
  red[t] = s; __syncthreads();
  for (int off = 128; off > 0; off >>= 1) {
    if (t < off) red[t] += red[t + off];
    __syncthreads();
  }
  if (t == 0) {
    const float lse = logf(red[0]) + mx;
    const float nll = lse - row[targets[n]];
    atomicAdd(loss, nll * (1.f / Nc));
  }
}

// ---------------------------------------------------------------------------
extern "C" void kernel_launch(void* const* d_in, const int* in_sizes, int n_in,
                              void* d_out, int out_size, void* d_ws, size_t ws_size,
                              hipStream_t stream) {
  (void)in_sizes; (void)n_in; (void)out_size; (void)ws_size;

  const int*   ids     = (const int*)d_in[0];
  const int*   tgt     = (const int*)d_in[1];
  const float* tok_emb = (const float*)d_in[2];
  const float* pos_emb = (const float*)d_in[3];
  const float* ln1_g   = (const float*)d_in[4];
  const float* ln1_b   = (const float*)d_in[5];
  const float* qkv_w   = (const float*)d_in[6];
  const float* qkv_b   = (const float*)d_in[7];
  const float* proj_w  = (const float*)d_in[8];
  const float* proj_b  = (const float*)d_in[9];
  const float* ln2_g   = (const float*)d_in[10];
  const float* ln2_b   = (const float*)d_in[11];
  const float* fc1_w   = (const float*)d_in[12];
  const float* fc1_b   = (const float*)d_in[13];
  const float* fc2_w   = (const float*)d_in[14];
  const float* fc2_b   = (const float*)d_in[15];
  const float* lnf_g   = (const float*)d_in[16];
  const float* lnf_b   = (const float*)d_in[17];

  // workspace layout (bytes)
  char* ws = (char*)d_ws;
  float*  x     = (float*) (ws);                           // N*D f32      8 MB
  __bf16* h16   = (__bf16*)(ws + ((size_t)8  << 20));      // N*D bf16     4 MB
  __bf16* qkv16 = (__bf16*)(ws + ((size_t)12 << 20));      // N*3D bf16   12 MB
  __bf16* y16   = (__bf16*)(ws + ((size_t)24 << 20));      // N*D bf16     4 MB
  __bf16* m16   = (__bf16*)(ws + ((size_t)28 << 20));      // N*DFF bf16  16 MB

  float* logits = (float*)d_out;                 // N*V fp32
  float* loss   = logits + (size_t)Nc * Vc;      // final scalar

  embed_kernel<<<Nc, 256, 0, stream>>>(ids, tok_emb, pos_emb, x);

  for (int l = 0; l < Lc; ++l) {
    ln_kernel<<<Nc, 256, 0, stream>>>(x, ln1_g + (size_t)l * Dc, ln1_b + (size_t)l * Dc, h16);
    // qkv = ln1(x) @ qkv_w^T + qkv_b   -> bf16
    gemm_bf16_wmma<<<dim3(QKVc / 64, Nc / 128), 256, 0, stream>>>(
        h16, qkv_w + (size_t)l * QKVc * Dc, qkv_b + (size_t)l * QKVc,
        nullptr, nullptr, qkv16, Dc, QKVc, 0);
    // fused flash attention: one wave per (b, h, 16 q-rows)
    attn_kernel<<<Bc * Hc * (Tc / 16), 32, 0, stream>>>(qkv16, y16);
    // x = x + y @ proj_w^T + proj_b
    gemm_bf16_wmma<<<dim3(Dc / 64, Nc / 128), 256, 0, stream>>>(
        y16, proj_w + (size_t)l * Dc * Dc, proj_b + (size_t)l * Dc,
        x, x, nullptr, Dc, Dc, 0);
    ln_kernel<<<Nc, 256, 0, stream>>>(x, ln2_g + (size_t)l * Dc, ln2_b + (size_t)l * Dc, h16);
    // m = gelu(ln2(x) @ fc1_w^T + fc1_b) -> bf16
    gemm_bf16_wmma<<<dim3(DFFc / 64, Nc / 128), 256, 0, stream>>>(
        h16, fc1_w + (size_t)l * DFFc * Dc, fc1_b + (size_t)l * DFFc,
        nullptr, nullptr, m16, Dc, DFFc, 1);
    // x = x + m @ fc2_w^T + fc2_b
    gemm_bf16_wmma<<<dim3(Dc / 64, Nc / 128), 256, 0, stream>>>(
        m16, fc2_w + (size_t)l * Dc * DFFc, fc2_b + (size_t)l * Dc,
        x, x, nullptr, DFFc, Dc, 0);
  }

  ln_kernel<<<Nc, 256, 0, stream>>>(x, lnf_g, lnf_b, h16);
  // logits = lnf(x) @ tok_emb^T  (no bias) -> d_out fp32
  gemm_bf16_wmma<<<dim3(Vc / 64, Nc / 128), 256, 0, stream>>>(
      h16, tok_emb, nullptr, nullptr, logits, nullptr, Dc, Vc, 0);

  zero_kernel<<<1, 1, 0, stream>>>(loss);
  loss_kernel<<<Nc, 256, 0, stream>>>(logits, tgt, loss);
}
